// Affinity_25022479467063
// MI455X (gfx1250) — matrix operand order repaired and checked by reading, not compiled, not run on hardware
//
#include <hip/hip_runtime.h>
#include <hip/hip_bf16.h>
#include <math.h>

// Problem constants (match reference setup_inputs)
#define BB   4
#define CC   64
#define LL   4096
#define HID  128
#define OO   64

typedef _Float16 v16h __attribute__((ext_vector_type(16)));
typedef _Float16 v8h  __attribute__((ext_vector_type(8)));
typedef float    v8f  __attribute__((ext_vector_type(8)));

#define SCALE       10.0f
#define THRESH      1e-3f
#define LEAKY       0.01f
#define NORM_EPS    1e-12f
// 10 * log2(e): hardware v_exp_f32 is base-2, so fold the softmax scale and
// the fixed shift (see kernel 2 comment) into the exp2 argument at compile
// time. The argument is bounded in [-28.85, 0] (normalized dots), so the raw
// v_exp_f32 (__builtin_amdgcn_exp2f) is exact -- no subnormal guard needed.
#define SCALE_LOG2E 14.426950408889634f

// LDS exp-score tile: 16 rows x 4096 cols, row stride padded to 4100 floats so
// the two half-wave row groups (rows r and r+8) hit disjoint LDS bank ranges
// (8 * 4100 mod 64 == 32).
#define SC_STRIDE   4100
#define SC_FLOATS   (16 * SC_STRIDE)
// stats area: per-wave partial sums [8 waves][16 rows] + global 1/sum [16]
#define STAT_FLOATS (8 * 16 + 16)
#define SHMEM_BYTES ((SC_FLOATS + STAT_FLOATS) * sizeof(float))

// ---------------------------------------------------------------------------
// Kernel 1: fused conv1x1 -> LeakyReLU -> conv1x1 -> L2-normalize -> fp16,
// written transposed as [B, L, 64] so WMMA fragments load contiguously.
// gridDim.y selects query (0) vs key (1); weights are shared (self-attention).
// ---------------------------------------------------------------------------
__global__ __launch_bounds__(256) void proj_norm_kernel(
    const float* __restrict__ q_in,   // [B, C, L]
    const float* __restrict__ k_in,   // [B, C, L]
    const float* __restrict__ W1,     // [128, 64]
    const float* __restrict__ W2,     // [64, 128]
    _Float16* __restrict__ qT,        // [B, L, 64]
    _Float16* __restrict__ kT)        // [B, L, 64]
{
    __shared__ float sW1[HID * CC];   // 32 KB
    __shared__ float sW2[OO * HID];   // 32 KB
    for (int i = threadIdx.x; i < HID * CC; i += 256) sW1[i] = W1[i];
    for (int i = threadIdx.x; i < OO * HID; i += 256) sW2[i] = W2[i];
    __syncthreads();

    const float* x    = (blockIdx.y == 0) ? q_in : k_in;
    _Float16*    dstT = (blockIdx.y == 0) ? qT  : kT;

    const int col = blockIdx.x * 256 + threadIdx.x;  // over B*L columns
    const int b   = col >> 12;                       // L = 4096
    const int l   = col & (LL - 1);

    // Gather this column's 64 channels (stride-L loads, coalesced across lanes).
    float xi[CC];
    const float* xp = x + (size_t)b * CC * LL + l;
    #pragma unroll
    for (int c = 0; c < CC; ++c) xi[c] = xp[(size_t)c * LL];

    float acc[OO];
    #pragma unroll
    for (int o = 0; o < OO; ++o) acc[o] = 0.0f;

    for (int j = 0; j < HID; ++j) {
        float h = 0.0f;
        #pragma unroll
        for (int c = 0; c < CC; ++c) h = fmaf(sW1[j * CC + c], xi[c], h);
        h = (h >= 0.0f) ? h : LEAKY * h;
        #pragma unroll
        for (int o = 0; o < OO; ++o) acc[o] = fmaf(sW2[o * HID + j], h, acc[o]);
    }

    float ss = 0.0f;
    #pragma unroll
    for (int o = 0; o < OO; ++o) ss = fmaf(acc[o], acc[o], ss);
    const float inv = 1.0f / fmaxf(sqrtf(ss), NORM_EPS);

    _Float16* dst = dstT + (size_t)col * OO;  // 128B contiguous per thread
    #pragma unroll
    for (int o = 0; o < OO; ++o) dst[o] = (_Float16)(acc[o] * inv);
}

// ---------------------------------------------------------------------------
// Kernel 2: affinity. One workgroup = one (batch, 16-row) stripe.
// Scores s = 10 * dot(q_hat, k_hat) are bounded in [-10, 10] because q,k are
// L2-normalized, and softmax is shift-invariant, so we use the FIXED shift
// exp(s - 10) = exp2(d*10*log2e - 10*log2e): no max pass, no rescaling, no
// over/underflow, and one FMA feeding a raw v_exp_f32 per element.
//  Pass 1: 16x16 tiles via v_wmma_f32_16x16x32_f16 (K=64 -> 2 WMMAs),
//          exp values stored in the 256KB LDS stripe (only possible with
//          CDNA5's 320KB/WGP LDS), per-row running sums in registers.
//  Reduce: wave32 shfl_xor add tree + cross-wave LDS combine -> 1/sum.
//  Pass 2: w = e * rinv, threshold, coalesced float4 stores (the only HBM
//          write: 268 MB = the bandwidth roofline floor at 23.3 TB/s).
// ---------------------------------------------------------------------------
__global__ __launch_bounds__(256) void affinity_kernel(
    const _Float16* __restrict__ qT,  // [B, L, 64]
    const _Float16* __restrict__ kT,  // [B, L, 64]
    float* __restrict__ outw)         // [B, L, L]
{
    extern __shared__ float lds[];
    float* sc    = lds;                       // [16][SC_STRIDE] exp-scores
    float* wvsum = sc + SC_FLOATS;            // [8][16] per-wave partial sums
    float* grinv = wvsum + 8 * 16;            // [16] 1/sum per row

    const int b    = blockIdx.y;
    const int l0   = blockIdx.x * 16;
    const int tid  = threadIdx.x;
    const int w    = tid >> 5;        // wave 0..7
    const int lane = tid & 31;
    const int hh   = lane >> 4;       // half-wave 0/1
    const int lq   = lane & 15;

    const _Float16* qb = qT + ((size_t)b * LL + l0) * OO;
    const _Float16* kb = kT + (size_t)b * LL * OO;

    // A fragments (16x32 f16, ISA layout): row m = lq; K chunks at 8*hh, +16.
    union { v16h v; v8h h[2]; } a0, a1;
    {
        const _Float16* arow = qb + lq * OO;
        const int kb8 = hh * 8;
        a0.h[0] = *(const v8h*)(arow + kb8);
        a0.h[1] = *(const v8h*)(arow + 16 + kb8);
        a1.h[0] = *(const v8h*)(arow + 32 + kb8);
        a1.h[1] = *(const v8h*)(arow + 48 + kb8);
    }

    float rsum[8];
    #pragma unroll
    for (int r = 0; r < 8; ++r) rsum[r] = 0.0f;

    // Pass 1: 32 tiles/wave * 8 waves = 256 16x16 tiles covering [16 x 4096].
    for (int t = 0; t < 32; ++t) {
        const int n0 = (t * 8 + w) * 16;
        // B fragments (32x16 f16): column n = lq; K chunks at 8*hh, +16.
        const _Float16* brow = kb + (size_t)(n0 + lq) * OO + hh * 8;
        union { v16h v; v8h h[2]; } b0, b1;
        b0.h[0] = *(const v8h*)(brow);
        b0.h[1] = *(const v8h*)(brow + 16);
        b1.h[0] = *(const v8h*)(brow + 32);
        b1.h[1] = *(const v8h*)(brow + 48);

        // Prefetch next iteration's k tile (global_prefetch_b8).
        if (t < 31) __builtin_prefetch(brow + 8 * 16 * OO, 0, 1);

        v8f c = {};
        c = __builtin_amdgcn_wmma_f32_16x16x32_f16(false, a0.v, false, b0.v,
                                                   (short)0, c, false, false);
        c = __builtin_amdgcn_wmma_f32_16x16x32_f16(false, a1.v, false, b1.v,
                                                   (short)0, c, false, false);

        const int ncol = n0 + lq;
        #pragma unroll
        for (int r = 0; r < 8; ++r) {
            // e = 2^(d*10*log2e - 10*log2e) = exp(10*d - 10) in [2^-28.85, 1];
            // argument >> -126 so the raw v_exp_f32 needs no subnormal guard.
            const float e   = __builtin_amdgcn_exp2f(
                                  fmaf(c[r], SCALE_LOG2E, -SCALE_LOG2E));
            const int   row = hh * 8 + r;          // C/D layout: half picks rows
            sc[row * SC_STRIDE + ncol] = e;
            rsum[r] += e;
        }
    }

    // Sum across the 16 lanes of each half-wave (wave32 shuffles).
    #pragma unroll
    for (int d = 1; d < 16; d <<= 1) {
        #pragma unroll
        for (int r = 0; r < 8; ++r) rsum[r] += __shfl_xor(rsum[r], d, 32);
    }
    if (lq == 0) {
        #pragma unroll
        for (int r = 0; r < 8; ++r) wvsum[w * 16 + hh * 8 + r] = rsum[r];
    }
    __syncthreads();

    // Combine the 8 per-wave partials into per-row reciprocal sums.
    if (tid < 16) {
        float s = 0.0f;
        #pragma unroll
        for (int ww = 0; ww < 8; ++ww) s += wvsum[ww * 16 + tid];
        grinv[tid] = 1.0f / s;
    }
    __syncthreads();

    // Pass 2: normalize + threshold + streaming store. 16*4096 floats / 256
    // threads = 64 float4 per thread, fully coalesced along columns.
    float* outb = outw + ((size_t)b * LL + l0) * LL;
    for (int i = 0; i < 64; ++i) {
        const int idx4 = i * 256 + tid;
        const int row  = idx4 >> 10;             // 1024 float4 per row
        const int c4   = (idx4 & 1023) * 4;
        const float rinv = grinv[row];
        const float4 ev = *(const float4*)&sc[row * SC_STRIDE + c4];
        float4 wv;
        wv.x = ev.x * rinv;
        wv.y = ev.y * rinv;
        wv.z = ev.z * rinv;
        wv.w = ev.w * rinv;
        wv.x = (wv.x > THRESH) ? wv.x : 0.0f;
        wv.y = (wv.y > THRESH) ? wv.y : 0.0f;
        wv.z = (wv.z > THRESH) ? wv.z : 0.0f;
        wv.w = (wv.w > THRESH) ? wv.w : 0.0f;
        *(float4*)&outb[(size_t)row * LL + c4] = wv;
    }
}

extern "C" void kernel_launch(void* const* d_in, const int* in_sizes, int n_in,
                              void* d_out, int out_size, void* d_ws, size_t ws_size,
                              hipStream_t stream) {
    const float* query = (const float*)d_in[0];
    const float* key   = (const float*)d_in[1];
    const float* W1    = (const float*)d_in[2];
    const float* W2    = (const float*)d_in[3];
    float* out = (float*)d_out;

    // Workspace: qT, kT as [B, L, 64] fp16 (2 MB each).
    _Float16* qT = (_Float16*)d_ws;
    _Float16* kT = qT + (size_t)BB * LL * OO;

    // Allow ~257 KB of dynamic LDS (gfx1250 WGP has 320 KB).
    (void)hipFuncSetAttribute((const void*)affinity_kernel,
                              hipFuncAttributeMaxDynamicSharedMemorySize,
                              (int)SHMEM_BYTES);

    // Kernel 1: projection + L2 norm for query and key (grid.y selects input).
    dim3 g1((BB * LL) / 256, 2);
    proj_norm_kernel<<<g1, 256, 0, stream>>>(query, key, W1, W2, qT, kT);

    // Kernel 2: WMMA Gram matrix + LDS-resident fixed-shift softmax + threshold.
    dim3 g2(LL / 16, BB);
    affinity_kernel<<<g2, 256, SHMEM_BYTES, stream>>>(qT, kT, out);
}
